// CausalSelfAttention_62766652063916
// MI455X (gfx1250) — matrix operand order repaired and checked by reading, not compiled, not run on hardware
//
#include <hip/hip_runtime.h>

typedef __attribute__((ext_vector_type(16))) _Float16 v16h;
typedef __attribute__((ext_vector_type(8)))  _Float16 v8h;
typedef __attribute__((ext_vector_type(8)))  float    v8f;
typedef __attribute__((ext_vector_type(4)))  float    v4f;
typedef __attribute__((ext_vector_type(4)))  unsigned int u32x4;
typedef __attribute__((ext_vector_type(8)))  int      i32x8;
typedef __attribute__((ext_vector_type(4)))  int      i32x4;

#define T_SEQ   4096
#define C_DIM   768
#define H_HEADS 12
#define D_HEAD  64
#define LDQKV   2304   // 3*C

// ---------------------------------------------------------------------------
// Generic GEMM:  Out[M,N] = A[M,K] * W[K,N] + bias[N]
// A may be f32 (x) or f16 (attention output); W/bias are f32; Out f16 or f32.
// Block: 256 threads = 8 waves; tile 128x256, BK=32.
// Wave grid 2x4, each wave owns 64x64 => 4x4 tiles of v_wmma_f32_16x16x32_f16
// (16 WMMAs per k-step, every fragment reused 4x => max matrix-op density).
// ---------------------------------------------------------------------------
template <typename TA, typename TO>
__global__ __launch_bounds__(256) void gemm_bias_wmma(
    const TA* __restrict__ A, const float* __restrict__ W,
    const float* __restrict__ bias, TO* __restrict__ Out,
    int M, int N, int K) {
  __shared__ __align__(128) _Float16 sA[128 * 32];   // row-major [m][k]
  __shared__ __align__(128) _Float16 sB[256 * 32];   // col-major [n][k]

  const int tid  = threadIdx.x;
  const int lane = tid & 31;
  const int wave = tid >> 5;
  const int lo   = lane & 15;
  const int hi   = lane >> 4;
  const int m0   = blockIdx.y * 128;
  const int n0   = blockIdx.x * 256;
  const int wm   = wave >> 2;   // 0..1  (64-row slab)
  const int wn   = wave & 3;    // 0..3  (64-col slab)

  v8f acc[4][4];
  v8f vzero = {0.f, 0.f, 0.f, 0.f, 0.f, 0.f, 0.f, 0.f};
#pragma unroll
  for (int i = 0; i < 4; i++)
#pragma unroll
    for (int j = 0; j < 4; j++) acc[i][j] = vzero;

  for (int k0 = 0; k0 < K; k0 += 32) {
    // ---- stage A tile (128x32) -> f16 LDS, 16 elements per thread ----
    {
      const int row  = tid >> 1;
      const int kOff = (tid & 1) * 16;
      const TA* src  = A + (size_t)(m0 + row) * K + k0 + kOff;
      _Float16* dst  = &sA[row * 32 + kOff];
      if constexpr (sizeof(TA) == 4) {
        v4f x0 = *(const v4f*)(src + 0);
        v4f x1 = *(const v4f*)(src + 4);
        v4f x2 = *(const v4f*)(src + 8);
        v4f x3 = *(const v4f*)(src + 12);
        v8h h0, h1;
#pragma unroll
        for (int i = 0; i < 4; i++) {
          h0[i]     = (_Float16)x0[i];
          h0[i + 4] = (_Float16)x1[i];
          h1[i]     = (_Float16)x2[i];
          h1[i + 4] = (_Float16)x3[i];
        }
        *(v8h*)dst       = h0;
        *(v8h*)(dst + 8) = h1;
      } else {
        *(v8h*)dst       = *(const v8h*)src;
        *(v8h*)(dst + 8) = *(const v8h*)(src + 8);
      }
      if (k0 + 32 < K) __builtin_prefetch(src + 32, 0, 0);  // global_prefetch_b8
    }
    // ---- stage B tile (32x256) transposed -> LDS [n][k], 32 elems/thread ----
    {
      const int n = tid;                       // one column per thread
      const float* src = W + (size_t)k0 * N + n0 + n;
      v8h h[4];
#pragma unroll
      for (int c = 0; c < 4; c++)
#pragma unroll
        for (int i = 0; i < 8; i++)
          h[c][i] = (_Float16)src[(size_t)(c * 8 + i) * N];
      _Float16* dst = &sB[n * 32];
#pragma unroll
      for (int c = 0; c < 4; c++) *(v8h*)(dst + 8 * c) = h[c];
    }
    __syncthreads();

    // ---- fragments + 16 WMMAs ----
    v16h a[4], b[4];
#pragma unroll
    for (int mt = 0; mt < 4; mt++) {
      const _Float16* p = &sA[(wm * 64 + mt * 16 + lo) * 32 + hi * 8];
      *(v8h*)&a[mt]       = *(const v8h*)p;        // K chunk0 (8 halfs)
      *((v8h*)&a[mt] + 1) = *(const v8h*)(p + 16); // K chunk1
    }
#pragma unroll
    for (int nt = 0; nt < 4; nt++) {
      const _Float16* p = &sB[(wn * 64 + nt * 16 + lo) * 32 + hi * 8];
      *(v8h*)&b[nt]       = *(const v8h*)p;
      *((v8h*)&b[nt] + 1) = *(const v8h*)(p + 16);
    }
#pragma unroll
    for (int mt = 0; mt < 4; mt++)
#pragma unroll
      for (int nt = 0; nt < 4; nt++)
        acc[mt][nt] = __builtin_amdgcn_wmma_f32_16x16x32_f16(
            false, a[mt], false, b[nt], (short)0, acc[mt][nt], false, false);
    __syncthreads();
  }

  // ---- epilogue: bias add, store ----
#pragma unroll
  for (int mt = 0; mt < 4; mt++)
#pragma unroll
    for (int nt = 0; nt < 4; nt++) {
      const int col  = n0 + wn * 64 + nt * 16 + lo;
      const float bv = bias[col];
#pragma unroll
      for (int r = 0; r < 8; r++) {
        const int row = m0 + wm * 64 + mt * 16 + hi * 8 + r;
        Out[(size_t)row * N + col] = (TO)(acc[mt][nt][r] + bv);
      }
    }
}

// ---------------------------------------------------------------------------
// Flash attention forward. Block = (head, 128 query rows), 8 waves; each wave
// owns 16 query rows with private online softmax. K tile fetched via TDM.
// qkv layout: [T][2304] f16, q = cols 0..767, k = 768.., v = 1536.. ; head h
// occupies 64 contiguous cols. y: [T][768] f16.
// ---------------------------------------------------------------------------
__global__ __launch_bounds__(256) void attn_fwd(
    const _Float16* __restrict__ qkv, _Float16* __restrict__ y) {
  __shared__ __align__(128) _Float16 sK[64 * 64];    // row-major [key][d]
  __shared__ __align__(128) _Float16 sVt[64 * 64];   // transposed [d][key]
  __shared__ __align__(128) _Float16 sP[8][16 * 64]; // per-wave P scratch

  const int tid  = threadIdx.x;
  const int lane = tid & 31;
  const int wave = tid >> 5;
  const int lo   = lane & 15;
  const int hi   = lane >> 4;
  const int qb   = blockIdx.x;  // 0..31
  const int h    = blockIdx.y;  // 0..11
  const int qBase = qb * 128 + wave * 16;

  // Q fragments for this wave's 16-row strip (D=64 -> 2 k-steps)
  v16h qf[2];
  {
    const _Float16* qp = qkv + (size_t)(qBase + lo) * LDQKV + h * D_HEAD;
#pragma unroll
    for (int kd = 0; kd < 2; kd++) {
      const _Float16* p = qp + kd * 32 + hi * 8;
      *(v8h*)&qf[kd]       = *(const v8h*)p;
      *((v8h*)&qf[kd] + 1) = *(const v8h*)(p + 16);
    }
  }

  v8f vzero = {0.f, 0.f, 0.f, 0.f, 0.f, 0.f, 0.f, 0.f};
  v8f o[4];
#pragma unroll
  for (int i = 0; i < 4; i++) o[i] = vzero;
  float mrow[8], lrow[8];
#pragma unroll
  for (int r = 0; r < 8; r++) { mrow[r] = -3.0e38f; lrow[r] = 0.f; }

  const unsigned int sKoff = (unsigned int)(unsigned long long)(void*)&sK[0];
  const int nkb = 2 * qb + 2;

  for (int kb = 0; kb < nkb; kb++) {
    // ---- K tile via Tensor Data Mover: 64x64 f16, row stride 2304 elems ----
    if (wave == 0) {
      unsigned long long ga =
          (unsigned long long)(const void*)(qkv + (size_t)(kb * 64) * LDQKV +
                                            C_DIM + h * D_HEAD);
      u32x4 g0;
      g0[0] = 1u;                       // count=1, is_restore=0, gather=0
      g0[1] = sKoff;                    // lds_addr
      g0[2] = (unsigned int)ga;         // global_addr[31:0]
      g0[3] = (unsigned int)((ga >> 32) & 0x1FFFFFFull) | (2u << 30);  // type=2
      i32x8 g1;
      g1[0] = (1 << 16);                       // data_size = 2 bytes
      g1[1] = (int)((LDQKV & 0xFFFF) << 16);   // tensor_dim0 lo16 @63:48
      g1[2] = (int)((T_SEQ & 0xFFFF) << 16);   // tensor_dim1 lo16 @95:80
      g1[3] = (int)(64 << 16);                 // tile_dim0 = 64 @127:112
      g1[4] = 64;                              // tile_dim1 = 64 @143:128
      g1[5] = LDQKV;                           // tensor_dim0_stride lo32
      g1[6] = 0;
      g1[7] = 0;
      i32x4 gz = {0, 0, 0, 0};
#if __clang_major__ >= 23
      i32x8 gz8 = {0, 0, 0, 0, 0, 0, 0, 0};
      __builtin_amdgcn_tensor_load_to_lds(g0, g1, gz, gz, gz8, 0);
#else
      __builtin_amdgcn_tensor_load_to_lds(g0, g1, gz, gz, 0);
#endif
      __builtin_amdgcn_s_wait_tensorcnt(0);
    }
    // ---- V tile staged transposed by all 256 threads ----
    {
      const int key  = tid >> 2;
      const int dOff = (tid & 3) * 16;
      const _Float16* vp = qkv + (size_t)(kb * 64 + key) * LDQKV + 2 * C_DIM +
                           h * D_HEAD + dOff;
      v8h v0 = *(const v8h*)vp;
      v8h v1 = *(const v8h*)(vp + 8);
#pragma unroll
      for (int i = 0; i < 8; i++) {
        sVt[(dOff + i) * 64 + key]     = v0[i];
        sVt[(dOff + 8 + i) * 64 + key] = v1[i];
      }
    }
    __syncthreads();

    const bool active = (kb * 64) <= (qBase + 15);  // wave-uniform
    if (active) {
      // ---- S = (Q K^T) * 1/sqrt(D), causal mask ----
      v8f s[4];
#pragma unroll
      for (int nt = 0; nt < 4; nt++) {
        v16h bk[2];
#pragma unroll
        for (int kd = 0; kd < 2; kd++) {
          const _Float16* p = &sK[(nt * 16 + lo) * 64 + kd * 32 + hi * 8];
          *(v8h*)&bk[kd]       = *(const v8h*)p;
          *((v8h*)&bk[kd] + 1) = *(const v8h*)(p + 16);
        }
        v8f z = vzero;
        z = __builtin_amdgcn_wmma_f32_16x16x32_f16(false, qf[0], false, bk[0],
                                                   (short)0, z, false, false);
        z = __builtin_amdgcn_wmma_f32_16x16x32_f16(false, qf[1], false, bk[1],
                                                   (short)0, z, false, false);
        s[nt] = z;
      }
#pragma unroll
      for (int nt = 0; nt < 4; nt++)
#pragma unroll
        for (int r = 0; r < 8; r++) {
          const int qi = qBase + hi * 8 + r;
          const int ki = kb * 64 + nt * 16 + lo;
          float v = s[nt][r] * 0.125f;  // 1/sqrt(64)
          s[nt][r] = (ki > qi) ? -3.0e38f : v;
        }

      // ---- online softmax (rows live across lanes 0..15 of each half) ----
      float rmax[8];
#pragma unroll
      for (int r = 0; r < 8; r++) {
        float v = fmaxf(fmaxf(s[0][r], s[1][r]), fmaxf(s[2][r], s[3][r]));
        v = fmaxf(v, __shfl_xor(v, 1, 32));
        v = fmaxf(v, __shfl_xor(v, 2, 32));
        v = fmaxf(v, __shfl_xor(v, 4, 32));
        v = fmaxf(v, __shfl_xor(v, 8, 32));
        rmax[r] = v;
      }
      float alpha[8];
#pragma unroll
      for (int r = 0; r < 8; r++) {
        const float mn = fmaxf(mrow[r], rmax[r]);
        alpha[r] = __expf(mrow[r] - mn);
        mrow[r]  = mn;
      }
      float rs[8] = {};
#pragma unroll
      for (int nt = 0; nt < 4; nt++)
#pragma unroll
        for (int r = 0; r < 8; r++) {
          const float p = __expf(s[nt][r] - mrow[r]);
          rs[r] += p;
          sP[wave][(hi * 8 + r) * 64 + nt * 16 + lo] = (_Float16)p;
        }
#pragma unroll
      for (int r = 0; r < 8; r++) {
        float v = rs[r];
        v += __shfl_xor(v, 1, 32);
        v += __shfl_xor(v, 2, 32);
        v += __shfl_xor(v, 4, 32);
        v += __shfl_xor(v, 8, 32);
        lrow[r] = lrow[r] * alpha[r] + v;
      }
#pragma unroll
      for (int dt = 0; dt < 4; dt++)
#pragma unroll
        for (int r = 0; r < 8; r++) o[dt][r] *= alpha[r];

      // ---- O += P V  (P via LDS layout conversion to A-fragments) ----
      v16h pf[2];
#pragma unroll
      for (int ks = 0; ks < 2; ks++) {
        const _Float16* p = &sP[wave][lo * 64 + ks * 32 + hi * 8];
        *(v8h*)&pf[ks]       = *(const v8h*)p;
        *((v8h*)&pf[ks] + 1) = *(const v8h*)(p + 16);
      }
#pragma unroll
      for (int dt = 0; dt < 4; dt++) {
        v16h bv[2];
#pragma unroll
        for (int ks = 0; ks < 2; ks++) {
          const _Float16* p = &sVt[(dt * 16 + lo) * 64 + ks * 32 + hi * 8];
          *(v8h*)&bv[ks]       = *(const v8h*)p;
          *((v8h*)&bv[ks] + 1) = *(const v8h*)(p + 16);
        }
        o[dt] = __builtin_amdgcn_wmma_f32_16x16x32_f16(
            false, pf[0], false, bv[0], (short)0, o[dt], false, false);
        o[dt] = __builtin_amdgcn_wmma_f32_16x16x32_f16(
            false, pf[1], false, bv[1], (short)0, o[dt], false, false);
      }
    }
    __syncthreads();  // protect sK/sVt before next iteration
  }

  // ---- normalize and write y (f16) ----
#pragma unroll
  for (int dt = 0; dt < 4; dt++)
#pragma unroll
    for (int r = 0; r < 8; r++) {
      const int row = qBase + hi * 8 + r;
      y[(size_t)row * C_DIM + h * D_HEAD + dt * 16 + lo] =
          (_Float16)(o[dt][r] / lrow[r]);
    }
}

// ---------------------------------------------------------------------------
extern "C" void kernel_launch(void* const* d_in, const int* in_sizes, int n_in,
                              void* d_out, int out_size, void* d_ws,
                              size_t ws_size, hipStream_t stream) {
  (void)in_sizes; (void)n_in; (void)out_size; (void)ws_size;
  const float* x      = (const float*)d_in[0];
  // d_in[1] = mask: causality computed analytically, unused
  const float* W_qkv  = (const float*)d_in[2];
  const float* b_qkv  = (const float*)d_in[3];
  const float* W_proj = (const float*)d_in[4];
  const float* b_proj = (const float*)d_in[5];
  float* out = (float*)d_out;

  _Float16* qkv_h = (_Float16*)d_ws;                               // [T][2304]
  _Float16* y_h   = (_Float16*)((char*)d_ws +
                                (size_t)T_SEQ * LDQKV * sizeof(_Float16));

  // 1) qkv = x @ W_qkv + b_qkv  (f32 in, f16 out)
  gemm_bias_wmma<float, _Float16>
      <<<dim3(LDQKV / 256, T_SEQ / 128), 256, 0, stream>>>(
          x, W_qkv, b_qkv, qkv_h, T_SEQ, LDQKV, C_DIM);

  // 2) flash attention per (query-block, head)
  attn_fwd<<<dim3(T_SEQ / 128, H_HEADS), 256, 0, stream>>>(qkv_h, y_h);

  // 3) out = y @ W_proj + b_proj  (f16 in, f32 out)
  gemm_bias_wmma<_Float16, float>
      <<<dim3(C_DIM / 256, T_SEQ / 128), 256, 0, stream>>>(
          y_h, W_proj, b_proj, out, T_SEQ, C_DIM, C_DIM);
}
